// LSTM_29996051595631
// MI455X (gfx1250) — matrix operand order repaired and checked by reading, not compiled, not run on hardware
//
#include <hip/hip_runtime.h>
#include <hip/hip_bf16.h>
#include <math.h>

#define TSTEPS 1000
#define NBATCH 16
#define NB     32          // fused fwd+bwd batch rows
#define IN0    440
#define K0P    448         // IN0 padded to multiple of 32
#define HDIM   512
#define K1P    1024
#define NGATES 4
#define FUSED  2048        // 4 gates * 512
#define NWG    32          // persistent workgroups in scan kernel

typedef __attribute__((ext_vector_type(16))) __bf16 v16bf;
typedef __attribute__((ext_vector_type(8)))  __bf16 v8bf;
typedef __attribute__((ext_vector_type(8)))  float  v8f;
typedef __attribute__((ext_vector_type(4)))  int    i4v;

#define ASG __attribute__((address_space(1)))
#define ASL __attribute__((address_space(3)))

union ABf { v16bf v; v8bf h[2]; };

#if __has_builtin(__builtin_amdgcn_global_load_async_to_lds_b128)
#define HAVE_ASYNC_LDS 1
#else
#define HAVE_ASYNC_LDS 0
#endif

__device__ __forceinline__ void wait_asynccnt0() {
#if __has_builtin(__builtin_amdgcn_s_wait_asynccnt)
    __builtin_amdgcn_s_wait_asynccnt(0);
#else
    asm volatile("s_wait_asynccnt 0x0" ::: "memory");
#endif
}

// ---------------------------------------------------------------------------
// Utility: zero u32 words
// ---------------------------------------------------------------------------
__global__ void zero_u32(unsigned int* p, unsigned long long n) {
    unsigned long long i = (unsigned long long)blockIdx.x * 256 + threadIdx.x;
    if (i < n) p[i] = 0u;
}

// ---------------------------------------------------------------------------
// Pack layer-0 input x (T,16,440) fp32 -> Xpk0[row=t*32+r][k<448] bf16
// rows 0..15 forward x[t], rows 16..31 time-reversed x[T-1-t]
// ---------------------------------------------------------------------------
__global__ void pack_x0(const float* __restrict__ x, __bf16* __restrict__ Xpk) {
    size_t idx = (size_t)blockIdx.x * 256 + threadIdx.x;
    if (idx >= (size_t)TSTEPS * NB * K0P) return;
    int k = (int)(idx % K0P);
    size_t row = idx / K0P;
    int t = (int)(row >> 5), r = (int)(row & 31);
    float v = 0.f;
    if (k < IN0) {
        int tt = (r < NBATCH) ? t : (TSTEPS - 1 - t);
        int b  = (r < NBATCH) ? r : (r - NBATCH);
        v = x[((size_t)tt * NBATCH + b) * IN0 + k];
    }
    Xpk[idx] = (__bf16)v;
}

// ---------------------------------------------------------------------------
// Pack layer-1 input from layer-0 hidden states (bf16, slot t+1 = h_t)
// y0[t] = concat(h_f[t], h_b[t]=h[T-1-t,16:]) ; x2 = concat([y0, y0[::-1]], b)
// ---------------------------------------------------------------------------
__global__ void pack_x1(const __bf16* __restrict__ Hb0, __bf16* __restrict__ Xpk1) {
    size_t idx = (size_t)blockIdx.x * 256 + threadIdx.x;
    if (idx >= (size_t)TSTEPS * NB * K1P) return;
    int k = (int)(idx & 1023);
    size_t row = idx >> 10;
    int t = (int)(row >> 5), r = (int)(row & 31);
    int tt, rr, kk;
    if (k < HDIM) {
        kk = k;
        if (r < NBATCH) { tt = t;              rr = r; }
        else            { tt = TSTEPS - 1 - t; rr = r - NBATCH; }
    } else {
        kk = k - HDIM;
        if (r < NBATCH) { tt = TSTEPS - 1 - t; rr = r + NBATCH; }
        else            { tt = t;              rr = r; }
    }
    Xpk1[idx] = Hb0[((size_t)(tt + 1) * NB + rr) * HDIM + kk];
}

// ---------------------------------------------------------------------------
// Pack 4 gate weight matrices (each H x din, fp32) into WMMA B-fragment order:
// Wpk[ntile][ktile][lane][e] bf16, 32B contiguous per lane per (ntile,ktile).
// B[k][n] = W_g[h][k], fused n = g*512 + h.
// Lane layout (16-bit B 32x16): k = (lane>=16 ? 8 : 0) + (e&7) + (e>=8 ? 16 : 0), n = ntile*16+lane&15
// ---------------------------------------------------------------------------
__global__ void pack_wu(const float* __restrict__ Wf, const float* __restrict__ Wi,
                        const float* __restrict__ Wo, const float* __restrict__ Wc,
                        int din, int Kp, __bf16* __restrict__ Wpk) {
    int nkt = Kp >> 5;
    size_t total = (size_t)128 * nkt * 512;
    size_t idx = (size_t)blockIdx.x * 256 + threadIdx.x;
    if (idx >= total) return;
    int e    = (int)(idx & 15);
    int lane = (int)((idx >> 4) & 31);
    size_t rest = idx >> 9;
    int kt    = (int)(rest % nkt);
    int ntile = (int)(rest / nkt);
    int n = ntile * 16 + (lane & 15);
    int g = n >> 9, h = n & 511;
    int k = kt * 32 + ((lane >= 16) ? 8 : 0) + (e & 7) + ((e >= 8) ? 16 : 0);
    const float* Wg = (g == 0) ? Wf : (g == 1) ? Wi : (g == 2) ? Wo : Wc;
    float v = (k < din) ? Wg[(size_t)h * din + k] : 0.f;
    Wpk[idx] = (__bf16)v;
}

__global__ void pack_bias(const float* __restrict__ bf, const float* __restrict__ bi,
                          const float* __restrict__ bo, const float* __restrict__ bc,
                          float* __restrict__ biasF) {
    int idx = blockIdx.x * 256 + threadIdx.x;
    if (idx >= FUSED) return;
    int g = idx >> 9, h = idx & 511;
    const float* bg = (g == 0) ? bf : (g == 1) ? bi : (g == 2) ? bo : bc;
    biasF[idx] = bg[h];
}

// ---------------------------------------------------------------------------
// WMMA GEMM: G[g][t][r][h] = Xpk(32000 x Kp) * Wpk^T + bias  (bf16 -> fp32)
// grid = (1000, 32); block = 256 (8 waves); wave -> (m-tile, n-tile)
// ---------------------------------------------------------------------------
__global__ __launch_bounds__(256)
void gemm_gates(const __bf16* __restrict__ Xpk, const __bf16* __restrict__ Wpk,
                const float* __restrict__ biasF, float* __restrict__ G, int Kp) {
    const int lane = threadIdx.x & 31;
    const int wave = threadIdx.x >> 5;
    const int mt = wave & 1;
    const int nt = wave >> 1;
    const int m0 = blockIdx.x * 32 + mt * 16;
    const int ntile = blockIdx.y * 4 + nt;
    const int n0 = ntile * 16;
    const bool lo = lane < 16;
    const int nkt = Kp >> 5;

    const __bf16* arow  = Xpk + (size_t)(m0 + (lane & 15)) * Kp;
    const __bf16* bbase = Wpk + ((size_t)ntile * nkt) * 512 + lane * 16;

    v8f acc;
    const float bv = biasF[n0 + (lane & 15)];
#pragma unroll
    for (int i = 0; i < 8; ++i) acc[i] = bv;

    for (int k = 0; k < Kp; k += 32) {
        ABf ua;
        ua.h[0] = *(const v8bf*)(arow + k + (lo ? 0 : 8));
        ua.h[1] = *(const v8bf*)(arow + k + 16 + (lo ? 0 : 8));
        v16bf bf = *(const v16bf*)(bbase + (size_t)(k >> 5) * 512);
        acc = __builtin_amdgcn_wmma_f32_16x16x32_bf16(false, ua.v, false, bf,
                                                      (short)0, acc, false, false);
    }

    const int ccol = n0 + (lane & 15);
    const int g  = ccol >> 9;
    const int hc = ccol & 511;
    const int rbase = m0 + (lo ? 0 : 8);
    const size_t GSTR = (size_t)TSTEPS * NB * HDIM;
#pragma unroll
    for (int v = 0; v < 8; ++v) {
        int row = rbase + v;
        int t = row >> 5, r = row & 31;
        G[(size_t)g * GSTR + ((size_t)t * NB + r) * HDIM + hc] = acc[v];
    }
}

// ---------------------------------------------------------------------------
// Persistent recurrent scan. 32 workgroups, each owns 16 H columns.
// 8 waves = 4 gates x 2 m-tiles; U slice (64KB bf16) resident in LDS.
// h_{t-1} (32KB bf16) staged into LDS per step via async global->LDS engine.
// Grid-wide generation barrier per timestep.
// ---------------------------------------------------------------------------
__global__ __launch_bounds__(256)
void lstm_scan(const float* __restrict__ G, const __bf16* __restrict__ Upk,
               __bf16* __restrict__ Hb, float* __restrict__ Hf,
               int* __restrict__ bcnt, int* __restrict__ bgen) {
    extern __shared__ __align__(16) char smem[];
    __bf16* uld  = (__bf16*)smem;                      // 4*16*512 bf16 = 64 KB
    __bf16* hshm = uld + NGATES * 16 * 512;            // 32*512 bf16  = 32 KB
    __shared__ float gacc[NGATES][NB][16];             // 8 KB cross-wave tiles
    __shared__ float cst[NB][16];                      // 2 KB cell state

    const int wg   = blockIdx.x;
    const int tid  = threadIdx.x;
    const int lane = tid & 31;
    const int wave = tid >> 5;
    const int g    = wave >> 1;
    const int mt   = wave & 1;
    const bool lo  = lane < 16;

    // Preload this workgroup's U slice (all 4 gates, 16 columns, K=512)
    for (int gg = 0; gg < NGATES; ++gg) {
        const uint4* s = (const uint4*)(Upk + (size_t)(gg * NWG + wg) * (16 * 512));
        uint4* d = (uint4*)(uld + (size_t)gg * (16 * 512));
        for (int i = tid; i < 1024; i += 256) d[i] = s[i];
    }
    for (int i = tid; i < NB * 16; i += 256) ((float*)cst)[i] = 0.f;
    __syncthreads();

    const __bf16* ubase = uld + (size_t)g * (16 * 512) + lane * 16;
    const int mrow = mt * 16 + (lane & 15);
    const size_t GSTR = (size_t)TSTEPS * NB * HDIM;

    for (int t = 0; t < TSTEPS; ++t) {
        // ---- stage h_{t-1} (32x512 bf16, 32KB) into LDS ----
        const __bf16* hsrc = Hb + (size_t)t * NB * HDIM;   // slot t = h_{t-1}
#if HAVE_ASYNC_LDS
        for (int i = tid; i < 2048; i += 256) {
            __builtin_amdgcn_global_load_async_to_lds_b128(
                (ASG i4v*)(i4v*)(hsrc + (size_t)i * 8),
                (ASL i4v*)(i4v*)(hshm + (size_t)i * 8),
                0, 0);
        }
        wait_asynccnt0();
#else
        for (int i = tid; i < 2048; i += 256)
            ((uint4*)hshm)[i] = ((const uint4*)hsrc)[i];
#endif
        __syncthreads();

        // ---- Uh GEMM: acc = h_{t-1}(tile) x U(tile)^T, K = 512, from LDS ----
        const __bf16* arow = hshm + (size_t)mrow * HDIM;
        v8f acc;
#pragma unroll
        for (int i = 0; i < 8; ++i) acc[i] = 0.f;
#pragma unroll
        for (int kt = 0; kt < 16; ++kt) {
            const int k = kt * 32;
            ABf ua;
            ua.h[0] = *(const v8bf*)(arow + k + (lo ? 0 : 8));
            ua.h[1] = *(const v8bf*)(arow + k + 16 + (lo ? 0 : 8));
            v16bf bf = *(const v16bf*)(ubase + (size_t)kt * 512);
            acc = __builtin_amdgcn_wmma_f32_16x16x32_bf16(false, ua.v, false, bf,
                                                          (short)0, acc, false, false);
        }
        {
            const int rb = mt * 16 + (lo ? 0 : 8);
            const int cc = lane & 15;
#pragma unroll
            for (int v = 0; v < 8; ++v) gacc[g][rb + v][cc] = acc[v];
        }
        __syncthreads();

        // ---- elementwise LSTM cell update (512 elems, 2 per thread) ----
#pragma unroll
        for (int p = 0; p < 2; ++p) {
            int e = tid + p * 256;
            int r = e >> 4, c = e & 15;
            int hc = wg * 16 + c;
            size_t gi = ((size_t)t * NB + r) * HDIM + hc;
            float vf = G[0 * GSTR + gi] + gacc[0][r][c];
            float vi = G[1 * GSTR + gi] + gacc[1][r][c];
            float vo = G[2 * GSTR + gi] + gacc[2][r][c];
            float vc = G[3 * GSTR + gi] + gacc[3][r][c];
            float ft = 1.f / (1.f + __expf(-vf));
            float it = 1.f / (1.f + __expf(-vi));
            float ot = 1.f / (1.f + __expf(-vo));
            float ct = it * tanhf(vc) + ft * cst[r][c];
            cst[r][c] = ct;
            float ht = ot * tanhf(ct);
            Hb[((size_t)(t + 1) * NB + r) * HDIM + hc] = (__bf16)ht;
            if (Hf) Hf[((size_t)t * NB + r) * HDIM + hc] = ht;
            if (t + 1 < TSTEPS) {   // prefetch next step's gate pre-activations
                __builtin_prefetch(&G[0 * GSTR + gi + (size_t)NB * HDIM], 0, 1);
                __builtin_prefetch(&G[3 * GSTR + gi + (size_t)NB * HDIM], 0, 1);
            }
        }

        // ---- grid-wide generation barrier ----
        __syncthreads();
        __threadfence();
        if (tid == 0) {
            int cur = atomicAdd(bgen, 0);
            if (atomicAdd(bcnt, 1) == NWG - 1) {
                atomicExch(bcnt, 0);
                atomicAdd(bgen, 1);
            } else {
                while (atomicAdd(bgen, 0) == cur) { __builtin_amdgcn_s_sleep(1); }
            }
        }
        __syncthreads();
        __threadfence();
    }
}

// ---------------------------------------------------------------------------
// Final output: out[t][b][0:512]=h1_f[t,b]; out[t][b][512:1024]=h1[T-1-t,b+16]
// ---------------------------------------------------------------------------
__global__ void emit_out(const float* __restrict__ Hf1, float* __restrict__ out) {
    size_t idx = (size_t)blockIdx.x * 256 + threadIdx.x;
    if (idx >= (size_t)TSTEPS * NBATCH * 1024) return;
    int j = (int)(idx & 1023);
    int b = (int)((idx >> 10) & 15);
    int t = (int)(idx >> 14);
    float v;
    if (j < HDIM) v = Hf1[((size_t)t * NB + b) * HDIM + j];
    else          v = Hf1[((size_t)(TSTEPS - 1 - t) * NB + (b + NBATCH)) * HDIM + (j - HDIM)];
    out[idx] = v;
}

// ---------------------------------------------------------------------------
extern "C" void kernel_launch(void* const* d_in, const int* in_sizes, int n_in,
                              void* d_out, int out_size, void* d_ws, size_t ws_size,
                              hipStream_t stream) {
    (void)in_sizes; (void)n_in; (void)out_size; (void)ws_size;
    const float* x   = (const float*)d_in[0];
    const float* Wf0 = (const float*)d_in[1],  *bf0 = (const float*)d_in[2],  *Uf0 = (const float*)d_in[3];
    const float* Wi0 = (const float*)d_in[4],  *bi0 = (const float*)d_in[5],  *Ui0 = (const float*)d_in[6];
    const float* Wo0 = (const float*)d_in[7],  *bo0 = (const float*)d_in[8],  *Uo0 = (const float*)d_in[9];
    const float* Wc0 = (const float*)d_in[10], *bc0 = (const float*)d_in[11], *Uc0 = (const float*)d_in[12];
    const float* Wf1 = (const float*)d_in[13], *bf1 = (const float*)d_in[14], *Uf1 = (const float*)d_in[15];
    const float* Wi1 = (const float*)d_in[16], *bi1 = (const float*)d_in[17], *Ui1 = (const float*)d_in[18];
    const float* Wo1 = (const float*)d_in[19], *bo1 = (const float*)d_in[20], *Uo1 = (const float*)d_in[21];
    const float* Wc1 = (const float*)d_in[22], *bc1 = (const float*)d_in[23], *Uc1 = (const float*)d_in[24];

    char* ws = (char*)d_ws;
    size_t off = 0;
    auto carve = [&](size_t bytes) -> char* {
        off = (off + 255) & ~(size_t)255;
        char* p = ws + off;
        off += bytes;
        return p;
    };
    int*    bcnt   = (int*)carve(8);                     // [cnt, gen]
    float*  biasF0 = (float*)carve((size_t)FUSED * 4);
    float*  biasF1 = (float*)carve((size_t)FUSED * 4);
    __bf16* Xpk0   = (__bf16*)carve((size_t)TSTEPS * NB * K0P * 2);
    __bf16* Wpk0   = (__bf16*)carve((size_t)128 * (K0P / 32) * 512 * 2);
    __bf16* Upk0   = (__bf16*)carve((size_t)128 * 16 * 512 * 2);
    __bf16* Upk1   = (__bf16*)carve((size_t)128 * 16 * 512 * 2);
    __bf16* Xpk1   = (__bf16*)carve((size_t)TSTEPS * NB * K1P * 2);
    __bf16* Wpk1   = (__bf16*)carve((size_t)128 * (K1P / 32) * 512 * 2);
    float*  G      = (float*)carve((size_t)NGATES * TSTEPS * NB * HDIM * 4);
    __bf16* Hb0    = (__bf16*)carve((size_t)(TSTEPS + 1) * NB * HDIM * 2);
    __bf16* Hb1    = (__bf16*)carve((size_t)(TSTEPS + 1) * NB * HDIM * 2);
    float*  Hf1    = (float*)carve((size_t)TSTEPS * NB * HDIM * 4);

    // Init: zero h_{-1} slots and barrier counters
    zero_u32<<<32, 256, 0, stream>>>((unsigned int*)Hb0, (size_t)NB * HDIM / 2);
    zero_u32<<<32, 256, 0, stream>>>((unsigned int*)Hb1, (size_t)NB * HDIM / 2);
    zero_u32<<<1, 256, 0, stream>>>((unsigned int*)bcnt, 2);

    // Pack weights / biases / inputs
    pack_bias<<<8, 256, 0, stream>>>(bf0, bi0, bo0, bc0, biasF0);
    pack_bias<<<8, 256, 0, stream>>>(bf1, bi1, bo1, bc1, biasF1);
    {
        size_t n = (size_t)TSTEPS * NB * K0P;
        pack_x0<<<(unsigned)((n + 255) / 256), 256, 0, stream>>>(x, Xpk0);
    }
    {
        size_t n = (size_t)128 * (K0P / 32) * 512;
        pack_wu<<<(unsigned)((n + 255) / 256), 256, 0, stream>>>(Wf0, Wi0, Wo0, Wc0, IN0, K0P, Wpk0);
    }
    {
        size_t n = (size_t)128 * 16 * 512;
        pack_wu<<<(unsigned)((n + 255) / 256), 256, 0, stream>>>(Uf0, Ui0, Uo0, Uc0, HDIM, HDIM, Upk0);
        pack_wu<<<(unsigned)((n + 255) / 256), 256, 0, stream>>>(Uf1, Ui1, Uo1, Uc1, HDIM, HDIM, Upk1);
    }
    {
        size_t n = (size_t)128 * (K1P / 32) * 512;
        pack_wu<<<(unsigned)((n + 255) / 256), 256, 0, stream>>>(Wf1, Wi1, Wo1, Wc1, K1P, K1P, Wpk1);
    }

    dim3 ggrid(TSTEPS, FUSED / 64);

    // ---- Layer 0 ----
    gemm_gates<<<ggrid, 256, 0, stream>>>(Xpk0, Wpk0, biasF0, G, K0P);
    lstm_scan<<<NWG, 256, 98304, stream>>>(G, Upk0, Hb0, (float*)nullptr, bcnt, bcnt + 1);

    // ---- Layer 1 ----
    {
        size_t n = (size_t)TSTEPS * NB * K1P;
        pack_x1<<<(unsigned)((n + 255) / 256), 256, 0, stream>>>(Hb0, Xpk1);
    }
    gemm_gates<<<ggrid, 256, 0, stream>>>(Xpk1, Wpk1, biasF1, G, K1P);
    lstm_scan<<<NWG, 256, 98304, stream>>>(G, Upk1, Hb1, Hf1, bcnt, bcnt + 1);

    // ---- Output assembly ----
    {
        size_t n = (size_t)TSTEPS * NBATCH * 1024;
        emit_out<<<(unsigned)((n + 255) / 256), 256, 0, stream>>>(Hf1, (float*)d_out);
    }
}
